// Homo_estimator_32_7541962572273
// MI455X (gfx1250) — compile-verified, matrix operand order checked by reading, not run on hardware
//
#include <hip/hip_runtime.h>
#include <hip/hip_bf16.h>
#include <math.h>

// ---------------------------------------------------------------------------
// Types for CDNA5 WMMA (gfx1250, wave32)
// ---------------------------------------------------------------------------
typedef _Float16 v8h  __attribute__((ext_vector_type(8)));
typedef _Float16 v16h __attribute__((ext_vector_type(16)));
typedef float    v8f  __attribute__((ext_vector_type(8)));
typedef unsigned int u32x4 __attribute__((ext_vector_type(4)));
typedef int          i32x4 __attribute__((ext_vector_type(4)));
typedef int          i32x8 __attribute__((ext_vector_type(8)));

union HFrag { v16h v; v8h h[2]; };

#define WMMA_F16(A, B, C) \
  __builtin_amdgcn_wmma_f32_16x16x32_f16(false, (A), false, (B), (short)0, (C), false, false)

// Tensor Data Mover availability (device pass only; host pass takes fallback).
#if defined(__gfx1250__) && __has_builtin(__builtin_amdgcn_tensor_load_to_lds) && \
    __has_builtin(__builtin_amdgcn_s_wait_tensorcnt)
#define USE_TDM 1
#else
#define USE_TDM 0
#endif

#if USE_TDM
// Issue one 2D TDM tile load: global (f16, row-major, row stride = stride0
// elements) -> LDS at byte offset lds_addr. tile = tile_d1 rows x tile_d0 elems.
// D# bitfields per CDNA5 ISA 8.3/8.4 (group0: count/lds/global/type,
// group1: data_size=2B, dims, strides). Groups 2/3 zero (2D tensor).
__device__ __forceinline__ void tdm_load_2d(unsigned lds_addr,
                                            unsigned long long gaddr,
                                            unsigned tile_d0, unsigned tile_d1,
                                            unsigned tensor_d0, unsigned tensor_d1,
                                            unsigned stride0)
{
  u32x4 g0;
  g0[0] = 1u;                                           // count=1, user mode
  g0[1] = lds_addr;                                     // lds_addr [63:32]
  g0[2] = (unsigned)gaddr;                              // global_addr [95:64]
  g0[3] = (unsigned)((gaddr >> 32) & 0x1FFFFFFu) | (2u << 30);   // addr hi | type=2
  i32x8 g1;
  g1[0] = (int)(1u << 16);                              // data_size=1 (2 bytes)
  g1[1] = (int)((tensor_d0 & 0xFFFFu) << 16);           // tensor_dim0 [15:0]
  g1[2] = (int)((tensor_d0 >> 16) | ((tensor_d1 & 0xFFFFu) << 16));
  g1[3] = (int)((tensor_d1 >> 16) | (tile_d0 << 16));   // tile_dim0
  g1[4] = (int)(tile_d1 & 0xFFFFu);                     // tile_dim1 (tile_dim2=0)
  g1[5] = (int)stride0;                                 // tensor_dim0_stride lo
  g1[6] = 0;
  g1[7] = 0;
  i32x4 g2 = {0, 0, 0, 0};
  i32x4 g3 = {0, 0, 0, 0};
#if __clang_major__ >= 23
  i32x8 g4 = {0, 0, 0, 0, 0, 0, 0, 0};
  __builtin_amdgcn_tensor_load_to_lds(g0, g1, g2, g3, g4, 0);
#else
  __builtin_amdgcn_tensor_load_to_lds(g0, g1, g2, g3, 0);
#endif
}
#endif

// ---------------------------------------------------------------------------
// f32 -> f16 cast (weights + images)
// ---------------------------------------------------------------------------
__global__ void __launch_bounds__(256)
k_cast_f32_f16(const float* __restrict__ s, _Float16* __restrict__ d, int n) {
  for (int i = blockIdx.x * blockDim.x + threadIdx.x; i < n; i += gridDim.x * blockDim.x)
    d[i] = (_Float16)s[i];
}

// ---------------------------------------------------------------------------
// Implicit-GEMM convolution with WMMA.
//   X  : [B, Cin, H, W]            f16 (NCHW)
//   Wt : [Cout, Cin*KH*KW]         f16 (OIHW flattened == GEMM A)
//   Y  : otrans==0 : [B, Cout, Ho, Wo] f16 (NCHW)
//        otrans==1 : [B, Ho*Wo, Cout]  f16 (pixel-major, for TDM-fed GEMM)
// Block: 128 threads = 4 waves (2x2); tile M=64 (Cout) x N=64 (pixels), K step 32.
// ---------------------------------------------------------------------------
__global__ void __launch_bounds__(128)
k_conv_wmma(const _Float16* __restrict__ X, const _Float16* __restrict__ Wt,
            const float* __restrict__ bias, _Float16* __restrict__ Y,
            int Bn, int Cin, int H, int W, int Cout, int KH, int KW,
            int pad, int stride, int relu, int otrans)
{
  const int Ho   = (H + 2 * pad - KH) / stride + 1;
  const int Wo   = (W + 2 * pad - KW) / stride + 1;
  const int HWo  = Ho * Wo;
  const int Npix = Bn * HWo;
  const int Ktot = Cin * KH * KW;
  const int KK   = KH * KW;

  __shared__ __align__(16) _Float16 As[64][32];   // [m][k]  (weights)
  __shared__ __align__(16) _Float16 Bs[64][32];   // [n][k]  (im2col pixels)

  const int tid    = threadIdx.x;
  const int lane   = tid & 31;
  const int wave   = tid >> 5;
  const int wm     = wave >> 1;
  const int wn     = wave & 1;
  const int laneLo = lane & 15;
  const int laneHi = lane >> 4;

  const int m0 = blockIdx.y * 64;
  const int n0 = blockIdx.x * 64;

  v8f acc[2][2] = {};

  for (int k0 = 0; k0 < Ktot; k0 += 32) {
    // ---- load A tile (weights), coalesced along K ----
#pragma unroll
    for (int i = 0; i < 16; ++i) {
      int e  = i * 128 + tid;
      int m  = e >> 5;
      int k  = e & 31;
      int gm = m0 + m, gk = k0 + k;
      _Float16 v = (_Float16)0.f;
      if (gm < Cout && gk < Ktot) {
        v = Wt[gm * Ktot + gk];
        if (gk + 32 < Ktot) __builtin_prefetch(&Wt[gm * Ktot + gk + 32], 0, 0);
      }
      As[m][k] = v;
    }
    // ---- load B tile (on-the-fly im2col) ----
#pragma unroll
    for (int i = 0; i < 16; ++i) {
      int e  = i * 128 + tid;
      int nn = e >> 5;
      int k  = e & 31;
      int gn = n0 + nn, gk = k0 + k;
      _Float16 v = (_Float16)0.f;
      if (gn < Npix && gk < Ktot) {
        int b   = gn / HWo;
        int hw  = gn - b * HWo;
        int oy  = hw / Wo;
        int ox  = hw - oy * Wo;
        int ci  = gk / KK;
        int rem = gk - ci * KK;
        int ky  = rem / KW;
        int kx  = rem - ky * KW;
        int iy  = oy * stride - pad + ky;
        int ix  = ox * stride - pad + kx;
        if (iy >= 0 && iy < H && ix >= 0 && ix < W)
          v = X[(((size_t)b * Cin + ci) * H + iy) * W + ix];
      }
      Bs[nn][k] = v;
    }
    __syncthreads();

    HFrag a[2], bfr[2];
#pragma unroll
    for (int t = 0; t < 2; ++t) {
      const _Float16* pa = &As[wm * 32 + t * 16 + laneLo][0];
      a[t].h[0] = *(const v8h*)(pa + (laneHi << 3));
      a[t].h[1] = *(const v8h*)(pa + 16 + (laneHi << 3));
      const _Float16* pb = &Bs[wn * 32 + t * 16 + laneLo][0];
      bfr[t].h[0] = *(const v8h*)(pb + (laneHi << 3));
      bfr[t].h[1] = *(const v8h*)(pb + 16 + (laneHi << 3));
    }
#pragma unroll
    for (int mi = 0; mi < 2; ++mi)
#pragma unroll
      for (int ni = 0; ni < 2; ++ni)
        acc[mi][ni] = WMMA_F16(a[mi].v, bfr[ni].v, acc[mi][ni]);
    __syncthreads();
  }

  // ---- epilogue: bias (+relu), scatter ----
#pragma unroll
  for (int mi = 0; mi < 2; ++mi) {
#pragma unroll
    for (int ni = 0; ni < 2; ++ni) {
      int n = n0 + wn * 32 + ni * 16 + laneLo;
      if (n >= Npix) continue;
      int b  = n / HWo;
      int hw = n - b * HWo;
      int mbase = m0 + wm * 32 + mi * 16 + laneHi * 8;
#pragma unroll
      for (int vg = 0; vg < 8; ++vg) {
        int mm = mbase + vg;
        if (mm < Cout) {
          float v = acc[mi][ni][vg];
          if (bias) v += bias[mm];
          if (relu) v = fmaxf(v, 0.f);
          if (otrans)
            Y[((size_t)b * HWo + hw) * Cout + mm] = (_Float16)v;
          else
            Y[(size_t)b * Cout * HWo + (size_t)mm * HWo + hw] = (_Float16)v;
        }
      }
    }
  }
}

// ---------------------------------------------------------------------------
// Batched correlation GEMM with WMMA:  C[b] = (1/16) * F1[b] * F2[b]^T
//   F1,F2: [B, HW, D] f16 (pixel-major), C: [B, HW, HW] f32.  M=N=HW, K=D.
// K-slices of both operands are 2D tiles (64 rows x 32 K, row stride D) and
// are DMA'd into LDS by the Tensor Data Mover, double-buffered so the TDM
// fetch of slice k+1 overlaps the WMMAs on slice k (TENSORcnt + barrier).
// ---------------------------------------------------------------------------
__global__ void __launch_bounds__(128)
k_corr_wmma(const _Float16* __restrict__ F1, const _Float16* __restrict__ F2,
            float* __restrict__ C, int D, int HW, float scale)
{
  __shared__ __align__(16) _Float16 As[2][64][32];   // [buf][m][k]
  __shared__ __align__(16) _Float16 Bs[2][64][32];   // [buf][n][k]

  const int tid = threadIdx.x, lane = tid & 31, wave = tid >> 5;
  const int wm = wave >> 1, wn = wave & 1;
  const int laneLo = lane & 15, laneHi = lane >> 4;
  const int m0 = blockIdx.x * 64, n0 = blockIdx.y * 64;

  const _Float16* f1 = F1 + ((size_t)blockIdx.z * HW + m0) * D;  // 64 rows x D
  const _Float16* f2 = F2 + ((size_t)blockIdx.z * HW + n0) * D;

  v8f acc[2][2] = {};
  int buf = 0;

#if USE_TDM
  const unsigned ldsA0 = (unsigned)(size_t)(void*)&As[0][0][0];
  const unsigned ldsA1 = (unsigned)(size_t)(void*)&As[1][0][0];
  const unsigned ldsB0 = (unsigned)(size_t)(void*)&Bs[0][0][0];
  const unsigned ldsB1 = (unsigned)(size_t)(void*)&Bs[1][0][0];
  if (wave == 0) {
    tdm_load_2d(ldsA0, (unsigned long long)(size_t)(const void*)f1,
                32u, 64u, (unsigned)D, (unsigned)HW, (unsigned)D);
    tdm_load_2d(ldsB0, (unsigned long long)(size_t)(const void*)f2,
                32u, 64u, (unsigned)D, (unsigned)HW, (unsigned)D);
  }
#endif

  for (int k0 = 0; k0 < D; k0 += 32) {
#if USE_TDM
    if (wave == 0) __builtin_amdgcn_s_wait_tensorcnt(0);
    __syncthreads();                       // tile[buf] visible to all waves
    if (wave == 0 && k0 + 32 < D) {        // prefetch next K-slice into buf^1
      tdm_load_2d(buf ? ldsA0 : ldsA1,
                  (unsigned long long)(size_t)(const void*)(f1 + k0 + 32),
                  32u, 64u, (unsigned)D, (unsigned)HW, (unsigned)D);
      tdm_load_2d(buf ? ldsB0 : ldsB1,
                  (unsigned long long)(size_t)(const void*)(f2 + k0 + 32),
                  32u, 64u, (unsigned)D, (unsigned)HW, (unsigned)D);
    }
#else
#pragma unroll
    for (int i = 0; i < 16; ++i) {
      int e = i * 128 + tid;
      int x = e >> 5;           // 0..63 row
      int k = e & 31;           // 0..31 (contiguous along D -> coalesced)
      As[buf][x][k] = f1[(size_t)x * D + k0 + k];
      Bs[buf][x][k] = f2[(size_t)x * D + k0 + k];
    }
    __syncthreads();
#endif

    HFrag a[2], bfr[2];
#pragma unroll
    for (int t = 0; t < 2; ++t) {
      const _Float16* pa = &As[buf][wm * 32 + t * 16 + laneLo][0];
      a[t].h[0] = *(const v8h*)(pa + (laneHi << 3));
      a[t].h[1] = *(const v8h*)(pa + 16 + (laneHi << 3));
      const _Float16* pb = &Bs[buf][wn * 32 + t * 16 + laneLo][0];
      bfr[t].h[0] = *(const v8h*)(pb + (laneHi << 3));
      bfr[t].h[1] = *(const v8h*)(pb + 16 + (laneHi << 3));
    }
#pragma unroll
    for (int mi = 0; mi < 2; ++mi)
#pragma unroll
      for (int ni = 0; ni < 2; ++ni)
        acc[mi][ni] = WMMA_F16(a[mi].v, bfr[ni].v, acc[mi][ni]);

    buf ^= 1;
    __syncthreads();   // readers of old buf done before it is refilled
  }

  size_t cb = (size_t)blockIdx.z * HW * HW;
#pragma unroll
  for (int mi = 0; mi < 2; ++mi)
#pragma unroll
    for (int ni = 0; ni < 2; ++ni) {
      int n = n0 + wn * 32 + ni * 16 + laneLo;
      int mbase = m0 + wm * 32 + mi * 16 + laneHi * 8;
#pragma unroll
      for (int vg = 0; vg < 8; ++vg)
        C[cb + (size_t)(mbase + vg) * HW + n] = acc[mi][ni][vg] * scale;
    }
}

// ---------------------------------------------------------------------------
// Instance norm (affine=False), in place on fp16, optional fused relu.
// ---------------------------------------------------------------------------
__global__ void __launch_bounds__(256)
k_inorm(_Float16* __restrict__ X, int HW, int relu)
{
  __shared__ float ssum[256], ssq[256];
  _Float16* p = X + (size_t)blockIdx.x * HW;
  float s = 0.f, q = 0.f;
  for (int i = threadIdx.x; i < HW; i += 256) { float v = (float)p[i]; s += v; q += v * v; }
  ssum[threadIdx.x] = s; ssq[threadIdx.x] = q;
  __syncthreads();
  for (int o = 128; o > 0; o >>= 1) {
    if (threadIdx.x < o) { ssum[threadIdx.x] += ssum[threadIdx.x + o]; ssq[threadIdx.x] += ssq[threadIdx.x + o]; }
    __syncthreads();
  }
  float mean = ssum[0] / (float)HW;
  float var  = ssq[0] / (float)HW - mean * mean;
  float rstd = rsqrtf(var + 1e-5f);
  for (int i = threadIdx.x; i < HW; i += 256) {
    float v = ((float)p[i] - mean) * rstd;
    if (relu) v = fmaxf(v, 0.f);
    p[i] = (_Float16)v;
  }
}

// ---------------------------------------------------------------------------
// Group norm + relu, in place on fp16. grid = B*G blocks.
// ---------------------------------------------------------------------------
__global__ void __launch_bounds__(256)
k_gnorm_relu(_Float16* __restrict__ X, const float* __restrict__ gw,
             const float* __restrict__ gb, int Cc, int G, int HW)
{
  __shared__ float ssum[256], ssq[256];
  int b = blockIdx.x / G, g = blockIdx.x % G;
  int cg = Cc / G;
  _Float16* p = X + ((size_t)b * Cc + (size_t)g * cg) * HW;
  int cnt = cg * HW;
  float s = 0.f, q = 0.f;
  for (int i = threadIdx.x; i < cnt; i += 256) { float v = (float)p[i]; s += v; q += v * v; }
  ssum[threadIdx.x] = s; ssq[threadIdx.x] = q;
  __syncthreads();
  for (int o = 128; o > 0; o >>= 1) {
    if (threadIdx.x < o) { ssum[threadIdx.x] += ssum[threadIdx.x + o]; ssq[threadIdx.x] += ssq[threadIdx.x + o]; }
    __syncthreads();
  }
  float mean = ssum[0] / (float)cnt;
  float var  = ssq[0] / (float)cnt - mean * mean;
  float rstd = rsqrtf(var + 1e-5f);
  for (int i = threadIdx.x; i < cnt; i += 256) {
    int c = g * cg + i / HW;
    float v = ((float)p[i] - mean) * rstd * gw[c] + gb[c];
    p[i] = (_Float16)fmaxf(v, 0.f);
  }
}

// ---------------------------------------------------------------------------
// dst = relu(a + b), fp16 elementwise (resblock tail).
// ---------------------------------------------------------------------------
__global__ void __launch_bounds__(256)
k_add_relu(_Float16* dst, const _Float16* a, const _Float16* b, int n)
{
  for (int i = blockIdx.x * blockDim.x + threadIdx.x; i < n; i += gridDim.x * blockDim.x)
    dst[i] = (_Float16)fmaxf((float)a[i] + (float)b[i], 0.f);
}

// ---------------------------------------------------------------------------
// 2x2 max pool (fp16) / 2x2 avg pool (f32), stride 2.
// ---------------------------------------------------------------------------
__global__ void __launch_bounds__(256)
k_maxpool2_h(const _Float16* __restrict__ X, _Float16* __restrict__ Y, int M, int H, int W)
{
  int Ho = H >> 1, Wo = W >> 1;
  int total = M * Ho * Wo;
  for (int i = blockIdx.x * blockDim.x + threadIdx.x; i < total; i += gridDim.x * blockDim.x) {
    int m = i / (Ho * Wo), r = i % (Ho * Wo);
    int oy = r / Wo, ox = r % Wo;
    const _Float16* p = X + ((size_t)m * H + oy * 2) * W + ox * 2;
    float v = fmaxf(fmaxf((float)p[0], (float)p[1]), fmaxf((float)p[W], (float)p[W + 1]));
    Y[i] = (_Float16)v;
  }
}

__global__ void __launch_bounds__(256)
k_avgpool2_f(const float* __restrict__ X, float* __restrict__ Y, int M, int H, int W)
{
  int Ho = H >> 1, Wo = W >> 1;
  int total = M * Ho * Wo;
  for (int i = blockIdx.x * blockDim.x + threadIdx.x; i < total; i += gridDim.x * blockDim.x) {
    int m = i / (Ho * Wo), r = i % (Ho * Wo);
    int oy = r / Wo, ox = r % Wo;
    const float* p = X + ((size_t)m * H + oy * 2) * W + ox * 2;
    Y[i] = 0.25f * (p[0] + p[1] + p[W] + p[W + 1]);
  }
}

// ---------------------------------------------------------------------------
// Correlation lookup: 2 levels, radius 4, identity grid, zero padding.
// corr0: [B*1024, 32, 32] f32, corr1: [B*1024, 16, 16] f32.
// Writes decoder input [B, 164, 32, 32] f16 (162 corr channels + 2 zero flow).
// ---------------------------------------------------------------------------
__device__ __forceinline__ float bl_gather(const float* src, int dim, int xi, int yi) {
  if (xi < 0 || xi > dim - 1 || yi < 0 || yi > dim - 1) return 0.f;
  return src[yi * dim + xi];
}

__global__ void __launch_bounds__(192)
k_corr_lookup(const float* __restrict__ corr0, const float* __restrict__ corr1,
              _Float16* __restrict__ dec_in)
{
  int b  = blockIdx.x >> 10;
  int hw = blockIdx.x & 1023;
  int h0 = hw >> 5, w0 = hw & 31;
  int t  = threadIdx.x;
  if (t >= 164) return;
  float val = 0.f;
  if (t < 162) {
    int lvl = t / 81, tap = t % 81;
    float dx = (float)(tap % 9) - 4.f;
    float dy = (float)(tap / 9) - 4.f;
    float cx = (lvl ? 0.5f * (float)w0 : (float)w0) + dx;
    float cy = (lvl ? 0.5f * (float)h0 : (float)h0) + dy;
    const float* src = lvl ? (corr1 + (size_t)blockIdx.x * 256)
                           : (corr0 + (size_t)blockIdx.x * 1024);
    int dim = lvl ? 16 : 32;
    float x0f = floorf(cx), y0f = floorf(cy);
    float wx = cx - x0f, wy = cy - y0f;
    int x0 = (int)x0f, y0 = (int)y0f;
    float v00 = bl_gather(src, dim, x0, y0);
    float v10 = bl_gather(src, dim, x0 + 1, y0);
    float v01 = bl_gather(src, dim, x0, y0 + 1);
    float v11 = bl_gather(src, dim, x0 + 1, y0 + 1);
    val = v00 * (1.f - wx) * (1.f - wy) + v10 * wx * (1.f - wy)
        + v01 * (1.f - wx) * wy + v11 * wx * wy;
  }
  dec_in[((size_t)b * 164 + t) * 1024 + hw] = (_Float16)val;
}

// ---------------------------------------------------------------------------
// Final 1x1 conv 256->2 at 2x2 + transpose(0,2,3,1).reshape(-1,4,2) -> f32 out.
// ---------------------------------------------------------------------------
__global__ void __launch_bounds__(256)
k_final_out(const _Float16* __restrict__ X, const float* __restrict__ Wf,
            const float* __restrict__ bf, float* __restrict__ out)
{
  int b = blockIdx.x;
  int wave = threadIdx.x >> 5, lane = threadIdx.x & 31;
  int pos = wave >> 1, c = wave & 1;
  int h = pos >> 1, w = pos & 1;
  float s = 0.f;
  for (int ci = lane; ci < 256; ci += 32)
    s += (float)X[(((size_t)b * 256 + ci) * 2 + h) * 2 + w] * Wf[c * 256 + ci];
  for (int o = 16; o > 0; o >>= 1) s += __shfl_xor(s, o, 32);
  if (lane == 0) out[(b * 4 + pos) * 2 + c] = s + bf[c];
}

// ---------------------------------------------------------------------------
// Host orchestration
// ---------------------------------------------------------------------------
static inline dim3 gs(long long n) {
  long long g = (n + 255) / 256;
  if (g > 8192) g = 8192;
  if (g < 1) g = 1;
  return dim3((unsigned)g);
}

static void launch_conv(const _Float16* X, const _Float16* Wt, const float* bias,
                        _Float16* Y, int B, int Cin, int H, int W, int Cout,
                        int KH, int KW, int pad, int stride, int relu, int otrans,
                        hipStream_t s)
{
  int Ho = (H + 2 * pad - KH) / stride + 1;
  int Wo = (W + 2 * pad - KW) / stride + 1;
  int Npix = B * Ho * Wo;
  dim3 grid((Npix + 63) / 64, (Cout + 63) / 64);
  k_conv_wmma<<<grid, 128, 0, s>>>(X, Wt, bias, Y, B, Cin, H, W, Cout, KH, KW,
                                   pad, stride, relu, otrans);
}

extern "C" void kernel_launch(void* const* d_in, const int* in_sizes, int n_in,
                              void* d_out, int out_size, void* d_ws, size_t ws_size,
                              hipStream_t stream)
{
  (void)n_in; (void)out_size; (void)ws_size;
  char* ws = (char*)d_ws;
  size_t off = 0;
  auto alloc = [&](size_t bytes) -> void* {
    void* p = ws + off;
    off = (off + bytes + 255) & ~(size_t)255;
    return p;
  };
  auto F32 = [&](int i) { return (const float*)d_in[i]; };

  // ---- pack conv weights to fp16 (OIHW flat == GEMM [Cout, K]) ----
  static const int wIdx[16] = {2, 4, 6, 8, 10, 12, 14, 16, 18, 20, 22, 24, 28, 32, 36, 40};
  _Float16* wp[16];
  for (int j = 0; j < 16; ++j) {
    int n = in_sizes[wIdx[j]];
    wp[j] = (_Float16*)alloc((size_t)n * sizeof(_Float16));
    k_cast_f32_f16<<<gs(n), 256, 0, stream>>>(F32(wIdx[j]), wp[j], n);
  }

  // ---- images -> fp16 ----
  int nimg = in_sizes[0];                       // 32*1*128*128
  _Float16* himg1 = (_Float16*)alloc((size_t)nimg * 2);
  _Float16* himg2 = (_Float16*)alloc((size_t)nimg * 2);
  k_cast_f32_f16<<<gs(nimg), 256, 0, stream>>>(F32(0), himg1, nimg);
  k_cast_f32_f16<<<gs(nimg), 256, 0, stream>>>(F32(1), himg2, nimg);

  // ---- activation buffers (fp16 unless noted) ----
  _Float16* A0 = (_Float16*)alloc((size_t)32 * 64 * 128 * 128 * 2);   // 67 MB
  _Float16* B0 = (_Float16*)alloc((size_t)32 * 64 * 64 * 64 * 2);
  _Float16* B1 = (_Float16*)alloc((size_t)32 * 64 * 64 * 64 * 2);
  _Float16* B2 = (_Float16*)alloc((size_t)32 * 64 * 64 * 64 * 2);
  _Float16* C0 = (_Float16*)alloc((size_t)32 * 64 * 32 * 32 * 2);
  _Float16* C1 = (_Float16*)alloc((size_t)32 * 96 * 32 * 32 * 2);
  _Float16* C2 = (_Float16*)alloc((size_t)32 * 96 * 32 * 32 * 2);
  _Float16* C3 = (_Float16*)alloc((size_t)32 * 96 * 32 * 32 * 2);
  _Float16* FM1 = (_Float16*)alloc((size_t)32 * 1024 * 256 * 2);      // [B,HW,D]
  _Float16* FM2 = (_Float16*)alloc((size_t)32 * 1024 * 256 * 2);      // [B,HW,D]
  float* CORR  = (float*)alloc((size_t)32 * 1024 * 1024 * 4);         // 134 MB
  float* CORR2 = (float*)alloc((size_t)32 * 1024 * 256 * 4);
  _Float16* DEC_IN = (_Float16*)alloc((size_t)32 * 164 * 1024 * 2);
  _Float16* D0 = (_Float16*)alloc((size_t)32 * 256 * 1024 * 2);
  _Float16* D1 = (_Float16*)alloc((size_t)32 * 256 * 1024 * 2);

  // ---- encoder (run twice, shared scratch) ----
  auto encode = [&](const _Float16* img, _Float16* fmap) {
    // conv7x7 1->64 @128, instance-norm + relu, maxpool -> 64x64
    launch_conv(img, wp[0], F32(3), A0, 32, 1, 128, 128, 64, 7, 7, 3, 1, 0, 0, stream);
    k_inorm<<<32 * 64, 256, 0, stream>>>(A0, 128 * 128, 1);
    k_maxpool2_h<<<gs((size_t)32 * 64 * 64 * 64), 256, 0, stream>>>(A0, B0, 32 * 64, 128, 128);
    // l1a / l1b (64->64 @64)
    for (int rb = 0; rb < 2; ++rb) {
      int wA = 1 + rb * 2, bA = 5 + rb * 4, wB = 2 + rb * 2, bB = 7 + rb * 4;
      launch_conv(B0, wp[wA], F32(bA), B1, 32, 64, 64, 64, 64, 3, 3, 1, 1, 0, 0, stream);
      k_inorm<<<32 * 64, 256, 0, stream>>>(B1, 4096, 1);
      launch_conv(B1, wp[wB], F32(bB), B2, 32, 64, 64, 64, 64, 3, 3, 1, 1, 0, 0, stream);
      k_inorm<<<32 * 64, 256, 0, stream>>>(B2, 4096, 0);
      k_add_relu<<<gs((size_t)32 * 64 * 4096), 256, 0, stream>>>(B0, B0, B2, 32 * 64 * 4096);
    }
    k_maxpool2_h<<<gs((size_t)32 * 64 * 1024), 256, 0, stream>>>(B0, C0, 32 * 64, 64, 64);
    // l2a (64->96 @32, 1x1 downsample on skip)
    launch_conv(C0, wp[5], F32(13), C1, 32, 64, 32, 32, 96, 3, 3, 1, 1, 0, 0, stream);
    k_inorm<<<32 * 96, 256, 0, stream>>>(C1, 1024, 1);
    launch_conv(C1, wp[6], F32(15), C2, 32, 96, 32, 32, 96, 3, 3, 1, 1, 0, 0, stream);
    k_inorm<<<32 * 96, 256, 0, stream>>>(C2, 1024, 0);
    launch_conv(C0, wp[9], F32(21), C3, 32, 64, 32, 32, 96, 1, 1, 0, 1, 0, 0, stream);
    k_inorm<<<32 * 96, 256, 0, stream>>>(C3, 1024, 0);
    k_add_relu<<<gs((size_t)32 * 96 * 1024), 256, 0, stream>>>(C1, C3, C2, 32 * 96 * 1024);
    // l2b (96->96 @32)
    launch_conv(C1, wp[7], F32(17), C2, 32, 96, 32, 32, 96, 3, 3, 1, 1, 0, 0, stream);
    k_inorm<<<32 * 96, 256, 0, stream>>>(C2, 1024, 1);
    launch_conv(C2, wp[8], F32(19), C3, 32, 96, 32, 32, 96, 3, 3, 1, 1, 0, 0, stream);
    k_inorm<<<32 * 96, 256, 0, stream>>>(C3, 1024, 0);
    k_add_relu<<<gs((size_t)32 * 96 * 1024), 256, 0, stream>>>(C1, C1, C3, 32 * 96 * 1024);
    // out 1x1 96->256, written pixel-major [B, HW, 256] for the TDM-fed GEMM
    launch_conv(C1, wp[10], F32(23), fmap, 32, 96, 32, 32, 256, 1, 1, 0, 1, 0, 1, stream);
  };
  encode(himg1, FM1);
  encode(himg2, FM2);

  // ---- correlation volume: per-batch GEMM 1024x1024x256, scale 1/sqrt(256) ----
  k_corr_wmma<<<dim3(16, 16, 32), 128, 0, stream>>>(FM1, FM2, CORR, 256, 1024, 0.0625f);
  k_avgpool2_f<<<gs((size_t)32 * 1024 * 256), 256, 0, stream>>>(CORR, CORR2, 32 * 1024, 32, 32);
  k_corr_lookup<<<32 * 1024, 192, 0, stream>>>(CORR, CORR2, DEC_IN);  // writes flow=0 too

  // ---- decoder: 4x [conv3x3 -> GN(32)+relu -> maxpool], then conv3x3+GN+relu ----
  const _Float16* x = DEC_IN;
  int cin = 164, S = 32;
  static const int dcB[4] = {25, 29, 33, 37};
  static const int dgW[4] = {26, 30, 34, 38};
  static const int dgB[4] = {27, 31, 35, 39};
  for (int i = 0; i < 4; ++i) {
    launch_conv(x, wp[11 + i], F32(dcB[i]), D0, 32, cin, S, S, 256, 3, 3, 1, 1, 0, 0, stream);
    k_gnorm_relu<<<32 * 32, 256, 0, stream>>>(D0, F32(dgW[i]), F32(dgB[i]), 256, 32, S * S);
    k_maxpool2_h<<<gs((size_t)32 * 256 * (S / 2) * (S / 2)), 256, 0, stream>>>(D0, D1, 32 * 256, S, S);
    x = D1; cin = 256; S >>= 1;
  }
  launch_conv(x, wp[15], F32(41), D0, 32, 256, 2, 2, 256, 3, 3, 1, 1, 0, 0, stream);
  k_gnorm_relu<<<32 * 32, 256, 0, stream>>>(D0, F32(42), F32(43), 256, 32, 4);

  // ---- final 1x1 conv 256->2 + transpose/reshape -> d_out f32 ----
  k_final_out<<<32, 256, 0, stream>>>(D0, F32(44), F32(45), (float*)d_out);
}
